// GAT_73237782331506
// MI455X (gfx1250) — compile-verified, matrix operand order checked by reading, not run on hardware
//
#include <hip/hip_runtime.h>
#include <hip/hip_bf16.h>

// ---------------------------------------------------------------------------
// 2-layer GATConv + sigmoid(z z^T) for MI455X (gfx1250, wave32, WMMA).
// All dense GEMMs use v_wmma_f32_16x16x32_bf16 with column-contiguous B
// fragments (weights are transposed at f32->bf16 conversion time), so every
// fragment load is a pair of global_load_b128.
// Edge softmax / aggregation via atomics (the memory-bound part).
// ---------------------------------------------------------------------------

#define NNODES 8192
#define NEDGES 262144
#define DIN    512
#define D1     128
#define D2     64
#define NH     4
#define NEG_SLOPE 0.2f

typedef __bf16 bf16_t;
typedef __attribute__((ext_vector_type(16))) __bf16 v16bf;
typedef __attribute__((ext_vector_type(8)))  float  v8f;

// ------------------------------ utility kernels ----------------------------

__global__ void fill_f32(float* __restrict__ p, float v, size_t n) {
    size_t i = (size_t)blockIdx.x * blockDim.x + threadIdx.x;
    if (i < n) p[i] = v;
}

__global__ void cvt_f32_bf16(const float* __restrict__ in, bf16_t* __restrict__ out, size_t n) {
    size_t i = (size_t)blockIdx.x * blockDim.x + threadIdx.x;
    if (i < n) out[i] = (bf16_t)in[i];
}

// out[c][r] = (bf16) in[r][c]   (in: rows x cols row-major)
__global__ void cvt_f32_bf16_T(const float* __restrict__ in, bf16_t* __restrict__ out,
                               int rows, int cols) {
    int idx = blockIdx.x * blockDim.x + threadIdx.x;
    if (idx >= rows * cols) return;
    int r = idx / cols, c = idx % cols;          // contiguous reads
    out[(size_t)c * rows + r] = (bf16_t)in[idx]; // scattered writes (tiny matrices)
}

// ------------------------------ WMMA GEMM ----------------------------------
// C[M,Nn] (f32, ldc) = A[M,K] (bf16, row-major, lda) * Bt^T
//   Bt holds B column-major: element B(k,n) = Bt[n*ldb + k]  -> contiguous in k.
//   ACT==1: sigmoid epilogue.
// One wave per 16x16 C tile.  K must be a multiple of 32.
//
// Fragment layouts (ISA 7.12.2, wave32):
//   A: lane(l16,half): row=l16; elems 0..7 = K(k0+8*half+i), 8..15 = K(k0+16+8*half+i)
//   B: lane(l16,half): col=l16; same K pattern
//   C: elem v -> (row = 8*half+v, col = l16)
template <int ACT>
__global__ void wmma_gemm_bf16(const bf16_t* __restrict__ A,
                               const bf16_t* __restrict__ Bt,
                               float* __restrict__ C,
                               int M, int Nn, int K,
                               int lda, int ldb, int ldc) {
    const int wavesPerBlock = blockDim.x >> 5;
    const int wave  = blockIdx.x * wavesPerBlock + (threadIdx.x >> 5);
    const int ntn   = Nn >> 4;
    const int tiles = (M >> 4) * ntn;
    if (wave >= tiles) return;
    const int mTile = wave / ntn;
    const int nTile = wave % ntn;

    const int lane = threadIdx.x & 31;
    const int half = lane >> 4;
    const int l16  = lane & 15;

    const bf16_t* ap0 = A  + (size_t)(mTile * 16 + l16) * lda + half * 8;
    const bf16_t* bp0 = Bt + (size_t)(nTile * 16 + l16) * ldb + half * 8;

    v8f acc = {};

    for (int k0 = 0; k0 < K; k0 += 32) {
        v16bf a, b;
        const bf16_t* ap = ap0 + k0;
        const bf16_t* bp = bp0 + k0;
        #pragma unroll
        for (int i = 0; i < 8; ++i) { a[i] = ap[i]; a[8 + i] = ap[16 + i]; }
        #pragma unroll
        for (int i = 0; i < 8; ++i) { b[i] = bp[i]; b[8 + i] = bp[16 + i]; }
        acc = __builtin_amdgcn_wmma_f32_16x16x32_bf16(
                  false, a, false, b, (short)0, acc, false, false);
    }

    float* cp = C + (size_t)(mTile * 16 + half * 8) * ldc + nTile * 16 + l16;
    #pragma unroll
    for (int v = 0; v < 8; ++v) {
        float x = acc[v];
        if (ACT == 1) x = 1.0f / (1.0f + expf(-x));
        cp[(size_t)v * ldc] = x;
    }
}

// ------------------------- attention coefficient ---------------------------
// el[n,h] = sum_d feat[n,h,d]*al[h,d] ; er likewise.  One thread per (n,h).
__global__ void attn_coeff(const float* __restrict__ feat,
                           const float* __restrict__ al,
                           const float* __restrict__ ar,
                           float* __restrict__ el, float* __restrict__ er, int D) {
    int idx = blockIdx.x * blockDim.x + threadIdx.x;
    if (idx >= NNODES * NH) return;
    int n = idx / NH, h = idx % NH;
    const float* f = feat + ((size_t)n * NH + h) * D;
    const float* a = al + h * D;
    const float* b = ar + h * D;
    float sl = 0.f, sr = 0.f;
    for (int d = 0; d < D; ++d) { float v = f[d]; sl += v * a[d]; sr += v * b[d]; }
    el[idx] = sl; er[idx] = sr;
}

// float atomic max via int/uint ordering trick (target init'd to -inf)
__device__ __forceinline__ void atomicMaxF(float* addr, float v) {
    if (v >= 0.f) atomicMax((int*)addr, __float_as_int(v));
    else          atomicMin((unsigned int*)addr, __float_as_uint(v));
}

// -------------------------- edge softmax passes ----------------------------

__global__ void edge_max(const int* __restrict__ src, const int* __restrict__ dst,
                         const float* __restrict__ el, const float* __restrict__ er,
                         float* __restrict__ emax) {
    int e = blockIdx.x * blockDim.x + threadIdx.x;
    if (e >= NEDGES) return;
    int s = src[e], t = dst[e];
    #pragma unroll
    for (int h = 0; h < NH; ++h) {
        float v = el[s * NH + h] + er[t * NH + h];
        v = v > 0.f ? v : NEG_SLOPE * v;
        atomicMaxF(&emax[t * NH + h], v);
    }
}

__global__ void edge_expsum(const int* __restrict__ src, const int* __restrict__ dst,
                            const float* __restrict__ el, const float* __restrict__ er,
                            const float* __restrict__ emax,
                            float* __restrict__ ee, float* __restrict__ esum) {
    int e = blockIdx.x * blockDim.x + threadIdx.x;
    if (e >= NEDGES) return;
    int s = src[e], t = dst[e];
    #pragma unroll
    for (int h = 0; h < NH; ++h) {
        float v = el[s * NH + h] + er[t * NH + h];
        v = v > 0.f ? v : NEG_SLOPE * v;
        float x = expf(v - emax[t * NH + h]);
        ee[(size_t)e * NH + h] = x;
        atomicAdd(&esum[t * NH + h], x);
    }
}

// acc[dst,h,d] += feat[src,h,d] * alpha ; one thread per (edge, d)
__global__ void edge_scatter(const int* __restrict__ src, const int* __restrict__ dst,
                             const float* __restrict__ feat,
                             const float* __restrict__ ee, const float* __restrict__ esum,
                             float* __restrict__ acc, int D) {
    size_t idx = (size_t)blockIdx.x * blockDim.x + threadIdx.x;
    if (idx >= (size_t)NEDGES * D) return;
    int    d = (int)(idx & (size_t)(D - 1));   // D is a power of two
    size_t e = idx / D;
    int s = src[e], t = dst[e];
    #pragma unroll
    for (int h = 0; h < NH; ++h) {
        float alpha = ee[e * NH + h] / esum[(size_t)t * NH + h];
        atomicAdd(&acc[((size_t)t * NH + h) * D + d],
                  feat[((size_t)s * NH + h) * D + d] * alpha);
    }
}

// out[n,d] = mean_h relu(acc[n,h,d] + bias[h,d]); also emits bf16 copy
__global__ void finalize_mean(const float* __restrict__ acc, const float* __restrict__ bias,
                              float* __restrict__ outf, bf16_t* __restrict__ outb, int D) {
    size_t idx = (size_t)blockIdx.x * blockDim.x + threadIdx.x;
    if (idx >= (size_t)NNODES * D) return;
    int n = (int)(idx / D), d = (int)(idx % D);
    float s = 0.f;
    #pragma unroll
    for (int h = 0; h < NH; ++h) {
        float v = acc[((size_t)n * NH + h) * D + d] + bias[h * D + d];
        s += v > 0.f ? v : 0.f;
    }
    float m = s * (1.0f / NH);
    outf[idx] = m;
    outb[idx] = (bf16_t)m;
}

// ------------------------------- launcher ----------------------------------

extern "C" void kernel_launch(void* const* d_in, const int* in_sizes, int n_in,
                              void* d_out, int out_size, void* d_ws, size_t ws_size,
                              hipStream_t stream) {
    const float* x   = (const float*)d_in[0];
    const int*   src = (const int*)  d_in[1];
    const int*   dst = (const int*)  d_in[2];
    const float* W1  = (const float*)d_in[3];
    const float* al1 = (const float*)d_in[4];
    const float* ar1 = (const float*)d_in[5];
    const float* b1  = (const float*)d_in[6];
    const float* W2  = (const float*)d_in[7];
    const float* al2 = (const float*)d_in[8];
    const float* ar2 = (const float*)d_in[9];
    const float* b2  = (const float*)d_in[10];
    float* out = (float*)d_out;

    // ---- workspace bump allocator (deterministic layout) ----
    char* base = (char*)d_ws;
    size_t off = 0;
    auto alloc = [&](size_t bytes) -> char* {
        char* p = base + off;
        off = (off + bytes + 255) & ~(size_t)255;
        return p;
    };
    // persistent across layers
    float*  hmean = (float*) alloc((size_t)NNODES * D1 * 4);
    bf16_t* hb    = (bf16_t*)alloc((size_t)NNODES * D1 * 2);
    float*  z     = (float*) alloc((size_t)NNODES * D2 * 4);
    bf16_t* zb    = (bf16_t*)alloc((size_t)NNODES * D2 * 2);
    size_t scratchBase = off;

    // layer-1 scratch
    bf16_t* xb    = (bf16_t*)alloc((size_t)NNODES * DIN * 2);
    bf16_t* w1t   = (bf16_t*)alloc((size_t)DIN * (NH * D1) * 2);   // W1^T, (NH*D1) x DIN
    float*  feat1 = (float*) alloc((size_t)NNODES * NH * D1 * 4);
    float*  el1   = (float*) alloc((size_t)NNODES * NH * 4);
    float*  er1   = (float*) alloc((size_t)NNODES * NH * 4);
    float*  emax1 = (float*) alloc((size_t)NNODES * NH * 4);
    float*  esum1 = (float*) alloc((size_t)NNODES * NH * 4);
    float*  ee1   = (float*) alloc((size_t)NEDGES * NH * 4);
    float*  acc1  = (float*) alloc((size_t)NNODES * NH * D1 * 4);

    // layer-2 scratch (aliases layer-1 scratch; layer-1 data is dead by then)
    off = scratchBase;
    bf16_t* w2t   = (bf16_t*)alloc((size_t)D1 * (NH * D2) * 2);    // W2^T, (NH*D2) x D1
    float*  feat2 = (float*) alloc((size_t)NNODES * NH * D2 * 4);
    float*  el2   = (float*) alloc((size_t)NNODES * NH * 4);
    float*  er2   = (float*) alloc((size_t)NNODES * NH * 4);
    float*  emax2 = (float*) alloc((size_t)NNODES * NH * 4);
    float*  esum2 = (float*) alloc((size_t)NNODES * NH * 4);
    float*  ee2   = (float*) alloc((size_t)NEDGES * NH * 4);
    float*  acc2  = (float*) alloc((size_t)NNODES * NH * D2 * 4);
    (void)ws_size; (void)n_in; (void)in_sizes; (void)out_size;

    auto cdiv = [](size_t a, size_t b) { return (unsigned)((a + b - 1) / b); };
    const int T = 256;

    // ======================= layer 1 =======================
    cvt_f32_bf16  <<<cdiv((size_t)NNODES * DIN, T), T, 0, stream>>>(x, xb, (size_t)NNODES * DIN);
    cvt_f32_bf16_T<<<cdiv((size_t)DIN * NH * D1, T), T, 0, stream>>>(W1, w1t, DIN, NH * D1);

    {   // feat1 = xb @ W1 : M=8192, N=512, K=512  (B from W1^T, contiguous)
        int tiles = (NNODES / 16) * ((NH * D1) / 16);
        wmma_gemm_bf16<0><<<cdiv(tiles, 8), 256, 0, stream>>>(
            xb, w1t, feat1, NNODES, NH * D1, DIN, DIN, DIN, NH * D1);
    }

    attn_coeff<<<cdiv((size_t)NNODES * NH, T), T, 0, stream>>>(feat1, al1, ar1, el1, er1, D1);
    fill_f32<<<cdiv((size_t)NNODES * NH, T), T, 0, stream>>>(emax1, -INFINITY, (size_t)NNODES * NH);
    fill_f32<<<cdiv((size_t)NNODES * NH, T), T, 0, stream>>>(esum1, 0.f, (size_t)NNODES * NH);
    fill_f32<<<cdiv((size_t)NNODES * NH * D1, T), T, 0, stream>>>(acc1, 0.f, (size_t)NNODES * NH * D1);

    edge_max   <<<cdiv(NEDGES, T), T, 0, stream>>>(src, dst, el1, er1, emax1);
    edge_expsum<<<cdiv(NEDGES, T), T, 0, stream>>>(src, dst, el1, er1, emax1, ee1, esum1);
    edge_scatter<<<cdiv((size_t)NEDGES * D1, T), T, 0, stream>>>(src, dst, feat1, ee1, esum1, acc1, D1);

    finalize_mean<<<cdiv((size_t)NNODES * D1, T), T, 0, stream>>>(acc1, b1, hmean, hb, D1);

    // ======================= layer 2 =======================
    cvt_f32_bf16_T<<<cdiv((size_t)D1 * NH * D2, T), T, 0, stream>>>(W2, w2t, D1, NH * D2);

    {   // feat2 = h @ W2 : M=8192, N=256, K=128  (B from W2^T, contiguous)
        int tiles = (NNODES / 16) * ((NH * D2) / 16);
        wmma_gemm_bf16<0><<<cdiv(tiles, 8), 256, 0, stream>>>(
            hb, w2t, feat2, NNODES, NH * D2, D1, D1, D1, NH * D2);
    }

    attn_coeff<<<cdiv((size_t)NNODES * NH, T), T, 0, stream>>>(feat2, al2, ar2, el2, er2, D2);
    fill_f32<<<cdiv((size_t)NNODES * NH, T), T, 0, stream>>>(emax2, -INFINITY, (size_t)NNODES * NH);
    fill_f32<<<cdiv((size_t)NNODES * NH, T), T, 0, stream>>>(esum2, 0.f, (size_t)NNODES * NH);
    fill_f32<<<cdiv((size_t)NNODES * NH * D2, T), T, 0, stream>>>(acc2, 0.f, (size_t)NNODES * NH * D2);

    edge_max   <<<cdiv(NEDGES, T), T, 0, stream>>>(src, dst, el2, er2, emax2);
    edge_expsum<<<cdiv(NEDGES, T), T, 0, stream>>>(src, dst, el2, er2, emax2, ee2, esum2);
    edge_scatter<<<cdiv((size_t)NEDGES * D2, T), T, 0, stream>>>(src, dst, feat2, ee2, esum2, acc2, D2);

    finalize_mean<<<cdiv((size_t)NNODES * D2, T), T, 0, stream>>>(acc2, b2, z, zb, D2);

    // =============== adjacency reconstruction ===============
    {   // out = sigmoid(zb @ zb^T) : M=N=8192, K=64 (zb is its own column-major B)
        int tiles = (NNODES / 16) * (NNODES / 16);
        wmma_gemm_bf16<1><<<cdiv(tiles, 8), 256, 0, stream>>>(
            zb, zb, out, NNODES, NNODES, D2, D2, D2, NNODES);
    }
}
